// Decoder_23536420782812
// MI455X (gfx1250) — matrix operand order repaired and checked by reading, not compiled, not run on hardware
//
#include <hip/hip_runtime.h>

// Shapes (fixed by the reference)
#define VOC 32000
#define DIM 512
#define HID 1024
#define BAT 32
#define SEQ 48
#define G4  4096   // 4*HID

typedef __attribute__((ext_vector_type(16))) __bf16 v16bf;
typedef __attribute__((ext_vector_type(8)))  __bf16 v8bf;
typedef __attribute__((ext_vector_type(8)))  float  v8f;

#define WMMA_BF16(a, b, c) \
  __builtin_amdgcn_wmma_f32_16x16x32_bf16(false, (a), false, (b), (short)0, (c), false, false)

__device__ __forceinline__ float sigmoidf_(float x) { return 1.0f / (1.0f + __expf(-x)); }

__device__ __forceinline__ v16bf pack16(v8bf lo, v8bf hi) {
  v16bf r;
#pragma unroll
  for (int e = 0; e < 8; ++e) { r[e] = lo[e]; r[8 + e] = hi[e]; }
  return r;
}

// ---------------------------------------------------------------------------
// A-tile: 16x32 bf16 tile from row-major bf16 [*, lda] at (m0,k0).
// lane = m, lane-half kh; element e: K = kh*16 + e (matches load_b)
// -> one contiguous 16-bf16 run per lane: 2x b128.
// ---------------------------------------------------------------------------
__device__ __forceinline__ v16bf load_a(const __bf16* __restrict__ A, int lda,
                                        int m0, int k0, int lane) {
  const __bf16* p = A + (size_t)(m0 + (lane & 15)) * lda + k0 + ((lane >> 4) << 4);
  return pack16(*(const v8bf*)p, *(const v8bf*)(p + 8));
}

// ---------------------------------------------------------------------------
// B-tile: B[k,n] = W[n,k] from row-major [N, ldw] at (n0,k0).
// lane = n, lane-half kh: K = kh*16 + e (one contiguous 16-element run).
// bf16 source: 2x b128.  f32 source (fallback): 4x b128 + cvt.
// ---------------------------------------------------------------------------
__device__ __forceinline__ v16bf load_b(const __bf16* __restrict__ W, int ldw,
                                        int n0, int k0, int lane) {
  const __bf16* p = W + (size_t)(n0 + (lane & 15)) * ldw + k0 + ((lane >> 4) << 4);
  return pack16(*(const v8bf*)p, *(const v8bf*)(p + 8));
}

__device__ __forceinline__ v16bf load_b(const float* __restrict__ W, int ldw,
                                        int n0, int k0, int lane) {
  const float* p = W + (size_t)(n0 + (lane & 15)) * ldw + k0 + ((lane >> 4) << 4);
  float4 f0 = *(const float4*)(p);
  float4 f1 = *(const float4*)(p + 4);
  float4 f2 = *(const float4*)(p + 8);
  float4 f3 = *(const float4*)(p + 12);
  v16bf b;
  b[0]  = (__bf16)f0.x; b[1]  = (__bf16)f0.y; b[2]  = (__bf16)f0.z; b[3]  = (__bf16)f0.w;
  b[4]  = (__bf16)f1.x; b[5]  = (__bf16)f1.y; b[6]  = (__bf16)f1.z; b[7]  = (__bf16)f1.w;
  b[8]  = (__bf16)f2.x; b[9]  = (__bf16)f2.y; b[10] = (__bf16)f2.z; b[11] = (__bf16)f2.w;
  b[12] = (__bf16)f3.x; b[13] = (__bf16)f3.y; b[14] = (__bf16)f3.z; b[15] = (__bf16)f3.w;
  return b;
}

// ---------------------------------------------------------------------------
// f32 -> bf16 bulk convert (8 elements / thread, b128 in / b128 out)
// ---------------------------------------------------------------------------
__global__ __launch_bounds__(256) void k_cvt_bf16(const float* __restrict__ src,
                                                  __bf16* __restrict__ dst, int n8) {
  int i = blockIdx.x * 256 + threadIdx.x;
  if (i >= n8) return;
  const float4* s = (const float4*)src + (size_t)i * 2;
  float4 f0 = s[0], f1 = s[1];
  v8bf d;
  d[0] = (__bf16)f0.x; d[1] = (__bf16)f0.y; d[2] = (__bf16)f0.z; d[3] = (__bf16)f0.w;
  d[4] = (__bf16)f1.x; d[5] = (__bf16)f1.y; d[6] = (__bf16)f1.z; d[7] = (__bf16)f1.w;
  *((v8bf*)dst + i) = d;
}

// ---------------------------------------------------------------------------
// K1: embedding gather -> bf16 X[t*B+b, :].  1536 rows * 64 chunks of 8.
// ---------------------------------------------------------------------------
__global__ __launch_bounds__(256) void k_embed(const int* __restrict__ tok,
                                               const float* __restrict__ emb,
                                               __bf16* __restrict__ X) {
  int i = blockIdx.x * 256 + threadIdx.x;       // [0, 98304)
  int m = i >> 6;                               // row (t*B+b)
  int q = i & 63;                               // 8-elem chunk
  int t = m >> 5, b = m & 31;
  int token = tok[b * SEQ + t];
  const float4* s = (const float4*)(emb + (size_t)token * DIM) + q * 2;
  float4 f0 = s[0], f1 = s[1];
  v8bf d;
  d[0] = (__bf16)f0.x; d[1] = (__bf16)f0.y; d[2] = (__bf16)f0.z; d[3] = (__bf16)f0.w;
  d[4] = (__bf16)f1.x; d[5] = (__bf16)f1.y; d[6] = (__bf16)f1.z; d[7] = (__bf16)f1.w;
  *((v8bf*)(X + (size_t)m * DIM) + q) = d;
}

// ---------------------------------------------------------------------------
// K2: hx0 = bf16(hidden_state[b, sent_len[b]-1, :]); cx0 = 0.
// ---------------------------------------------------------------------------
__global__ __launch_bounds__(256) void k_init(const float* __restrict__ hs_in,
                                              const int* __restrict__ slen,
                                              __bf16* __restrict__ h0,
                                              float* __restrict__ c0) {
  int i = blockIdx.x * 256 + threadIdx.x;       // [0, 32768)
  int b = i >> 10, h = i & 1023;
  int s = slen[b] - 1;
  h0[i] = (__bf16)hs_in[((size_t)b * SEQ + s) * HID + h];
  c0[i] = 0.0f;
}

// ---------------------------------------------------------------------------
// K3: x_gates = X @ W_ih^T + (b_ih+b_hh).  M=1536,N=4096,K=512.
// 32x64 per wave: 48*64 = 3072 waves -> 384 blocks.
// ---------------------------------------------------------------------------
template <typename TB>
__global__ __launch_bounds__(256) void k_gemm_xgates(const __bf16* __restrict__ X,
                                                     const TB* __restrict__ Wih,
                                                     const float* __restrict__ bih,
                                                     const float* __restrict__ bhh,
                                                     float* __restrict__ xg) {
  int w    = (blockIdx.x * 256 + threadIdx.x) >> 5;
  int lane = threadIdx.x & 31;
  int mg = w >> 6, ng = w & 63;                 // 48 x 64 groups
  int m0 = mg * 32, n0 = ng * 64;
  v8f acc[2][4] = {};
  for (int k0 = 0; k0 < DIM; k0 += 32) {
    v16bf a0 = load_a(X, DIM, m0, k0, lane);
    v16bf a1 = load_a(X, DIM, m0 + 16, k0, lane);
#pragma unroll
    for (int j = 0; j < 4; ++j) {
      v16bf b = load_b(Wih, DIM, n0 + j * 16, k0, lane);
      acc[0][j] = WMMA_BF16(a0, b, acc[0][j]);
      acc[1][j] = WMMA_BF16(a1, b, acc[1][j]);
    }
  }
  int cc = lane & 15, rb = (lane >> 4) << 3;
#pragma unroll
  for (int j = 0; j < 4; ++j) {
    int col   = n0 + j * 16 + cc;
    float bias = bih[col] + bhh[col];
#pragma unroll
    for (int mi = 0; mi < 2; ++mi) {
      int rbase = m0 + mi * 16 + rb;
#pragma unroll
      for (int r = 0; r < 8; ++r)
        xg[(size_t)(rbase + r) * G4 + col] = acc[mi][j][r] + bias;
    }
  }
}

// ---------------------------------------------------------------------------
// K4: recurrent step partial GEMM, split-K=4: gpart[kc] = Hprev @ W_hh^T |_Kslice
// M=32,N=4096. 16x16 tile/wave, 4*512 = 2048 waves -> 256 blocks.
// ---------------------------------------------------------------------------
template <typename TB>
__global__ __launch_bounds__(256) void k_gemm_step(const __bf16* __restrict__ Hprev,
                                                   const TB* __restrict__ Whh,
                                                   float* __restrict__ gpart) {
  int w    = (blockIdx.x * 256 + threadIdx.x) >> 5;
  int lane = threadIdx.x & 31;
  int kc   = w >> 9;                            // 0..3 K-slice
  int tile = w & 511;
  int mt = tile >> 8, nt = tile & 255;          // 2 x 256 tiles
  int m0 = mt * 16, n0 = nt * 16;
  v8f acc = {};
  int kbeg = kc * (HID / 4);
  for (int k0 = kbeg; k0 < kbeg + HID / 4; k0 += 32) {
    v16bf a = load_a(Hprev, HID, m0, k0, lane);
    v16bf b = load_b(Whh, HID, n0, k0, lane);
    acc = WMMA_BF16(a, b, acc);
  }
  float* gp = gpart + (size_t)kc * BAT * G4;
  int col   = n0 + (lane & 15);
  int rbase = m0 + ((lane >> 4) << 3);
#pragma unroll
  for (int r = 0; r < 8; ++r)
    gp[(size_t)(rbase + r) * G4 + col] = acc[r];
}

// ---------------------------------------------------------------------------
// K5: pointwise LSTM: sum 4 K-partials + x_gates[t]; c'=sig(f)c+sig(i)tanh(g);
// h'=sig(o)tanh(c') written as bf16 for the next GEMMs.
// ---------------------------------------------------------------------------
__global__ __launch_bounds__(256) void k_lstm_pointwise(const float* __restrict__ gpart,
                                                        const float* __restrict__ xg_t,
                                                        float* __restrict__ cbuf,
                                                        __bf16* __restrict__ hnext) {
  int idx = blockIdx.x * 256 + threadIdx.x;     // [0, 32768)
  int b = idx >> 10, h = idx & 1023;
  size_t base = (size_t)b * G4;
  float vi = xg_t[base + h];
  float vf = xg_t[base + HID + h];
  float vg = xg_t[base + 2 * HID + h];
  float vo = xg_t[base + 3 * HID + h];
#pragma unroll
  for (int kc = 0; kc < 4; ++kc) {
    const float* gp = gpart + (size_t)kc * BAT * G4 + base;
    vi += gp[h]; vf += gp[HID + h]; vg += gp[2 * HID + h]; vo += gp[3 * HID + h];
  }
  float ig = sigmoidf_(vi), fg = sigmoidf_(vf);
  float gg = tanhf(vg),     og = sigmoidf_(vo);
  float c  = fg * cbuf[idx] + ig * gg;
  cbuf[idx]  = c;
  hnext[idx] = (__bf16)(og * tanhf(c));
}

// ---------------------------------------------------------------------------
// K6: logits = Hs @ W_out^T + b_out with (t,b)->(b,t) remap.
// M=1536,N=32000,K=1024. 32x64 per wave: 48*500 = 24000 waves -> 3000 blocks.
// ---------------------------------------------------------------------------
template <typename TB>
__global__ __launch_bounds__(256) void k_gemm_out(const __bf16* __restrict__ Hs,
                                                  const TB* __restrict__ Wout,
                                                  const float* __restrict__ bout,
                                                  float* __restrict__ out) {
  int w    = (blockIdx.x * 256 + threadIdx.x) >> 5;
  int lane = threadIdx.x & 31;
  int mg = w / 500, ng = w % 500;
  int m0 = mg * 32, n0 = ng * 64;
  v8f acc[2][4] = {};
  for (int k0 = 0; k0 < HID; k0 += 32) {
    v16bf a0 = load_a(Hs, HID, m0, k0, lane);
    v16bf a1 = load_a(Hs, HID, m0 + 16, k0, lane);
#pragma unroll
    for (int j = 0; j < 4; ++j) {
      v16bf b = load_b(Wout, HID, n0 + j * 16, k0, lane);
      acc[0][j] = WMMA_BF16(a0, b, acc[0][j]);
      acc[1][j] = WMMA_BF16(a1, b, acc[1][j]);
    }
  }
  int cc = lane & 15, rb = (lane >> 4) << 3;
  float bb[4];
#pragma unroll
  for (int j = 0; j < 4; ++j) bb[j] = bout[n0 + j * 16 + cc];
#pragma unroll
  for (int mi = 0; mi < 2; ++mi) {
#pragma unroll
    for (int r = 0; r < 8; ++r) {
      int m = m0 + mi * 16 + rb + r;
      int t = m >> 5, b = m & 31;               // Hs rows ordered t*B+b
      float* orow = out + (size_t)(b * SEQ + t) * VOC + n0 + cc;
#pragma unroll
      for (int j = 0; j < 4; ++j)
        orow[j * 16] = acc[mi][j][r] + bb[j];
    }
  }
}

// ---------------------------------------------------------------------------
// K7: per-row argmax over V=32000; predicts (as float) after logits.
// ---------------------------------------------------------------------------
__global__ __launch_bounds__(256) void k_argmax(const float* __restrict__ logits,
                                                float* __restrict__ pred) {
  __shared__ float sv[256];
  __shared__ int   si[256];
  int row = blockIdx.x;
  const float* p = logits + (size_t)row * VOC;
  float best = -INFINITY; int bi = 0;
  for (int c = threadIdx.x; c < VOC; c += 256) {
    float v = p[c];
    if (v > best) { best = v; bi = c; }
  }
  sv[threadIdx.x] = best; si[threadIdx.x] = bi;
  __syncthreads();
  for (int s = 128; s > 0; s >>= 1) {
    if (threadIdx.x < s) {
      float v2 = sv[threadIdx.x + s]; int i2 = si[threadIdx.x + s];
      if (v2 > sv[threadIdx.x] || (v2 == sv[threadIdx.x] && i2 < si[threadIdx.x])) {
        sv[threadIdx.x] = v2; si[threadIdx.x] = i2;
      }
    }
    __syncthreads();
  }
  if (threadIdx.x == 0) pred[row] = (float)si[0];
}

// ---------------------------------------------------------------------------
extern "C" void kernel_launch(void* const* d_in, const int* in_sizes, int n_in,
                              void* d_out, int out_size, void* d_ws, size_t ws_size,
                              hipStream_t stream) {
  const int*   sent_inputs  = (const int*)d_in[0];
  const float* hidden_state = (const float*)d_in[1];
  const int*   sent_len     = (const int*)d_in[2];
  // d_in[3] = teacher_forcing_ratio (teacher forcing hard-coded true)
  const float* embedding    = (const float*)d_in[4];
  const float* W_ih         = (const float*)d_in[5];
  const float* W_hh         = (const float*)d_in[6];
  const float* b_ih         = (const float*)d_in[7];
  const float* b_hh         = (const float*)d_in[8];
  const float* W_out        = (const float*)d_in[9];
  const float* b_out        = (const float*)d_in[10];
  float* out = (float*)d_out;

  // Workspace carve-out (256B aligned chunks)
  size_t off = 0;
  auto carve = [&](size_t bytes) -> void* {
    void* p = (char*)d_ws + off;
    off += (bytes + 255) & ~(size_t)255;
    return p;
  };
  __bf16* Xb    = (__bf16*)carve((size_t)SEQ * BAT * DIM * 2);        // 1.6 MB
  float*  xg    = (float*) carve((size_t)SEQ * BAT * G4 * 4);         // 25.2 MB
  __bf16* hbuf  = (__bf16*)carve((size_t)(SEQ + 1) * BAT * HID * 2);  // 3.2 MB
  float*  cbuf  = (float*) carve((size_t)BAT * HID * 4);              // 0.13 MB
  float*  gpart = (float*) carve((size_t)4 * BAT * G4 * 4);           // 2.1 MB
  size_t base_end = off;
  __bf16* Wih_b  = (__bf16*)carve((size_t)G4 * DIM * 2);              // 4.2 MB
  __bf16* Whh_b  = (__bf16*)carve((size_t)G4 * HID * 2);              // 8.4 MB
  size_t small_end = off;
  __bf16* Wout_b = (__bf16*)carve((size_t)VOC * HID * 2);             // 65.5 MB
  size_t full_end = off;
  (void)base_end;

  const bool cvt_small = ws_size >= small_end;   // bf16 W_ih, W_hh
  const bool cvt_big   = ws_size >= full_end;    // + bf16 W_out

  k_embed<<<384, 256, 0, stream>>>(sent_inputs, embedding, Xb);
  k_init<<<128, 256, 0, stream>>>(hidden_state, sent_len, hbuf, cbuf);

  if (cvt_small) {
    k_cvt_bf16<<<(G4 * DIM / 8 + 255) / 256, 256, 0, stream>>>(W_ih, Wih_b, G4 * DIM / 8);
    k_cvt_bf16<<<(G4 * HID / 8 + 255) / 256, 256, 0, stream>>>(W_hh, Whh_b, G4 * HID / 8);
    k_gemm_xgates<__bf16><<<384, 256, 0, stream>>>(Xb, Wih_b, b_ih, b_hh, xg);
  } else {
    k_gemm_xgates<float><<<384, 256, 0, stream>>>(Xb, W_ih, b_ih, b_hh, xg);
  }
  if (cvt_big)
    k_cvt_bf16<<<(VOC * HID / 8 + 255) / 256, 256, 0, stream>>>(W_out, Wout_b, VOC * HID / 8);

  for (int t = 0; t < SEQ; ++t) {
    if (cvt_small)
      k_gemm_step<__bf16><<<256, 256, 0, stream>>>(hbuf + (size_t)t * BAT * HID, Whh_b, gpart);
    else
      k_gemm_step<float><<<256, 256, 0, stream>>>(hbuf + (size_t)t * BAT * HID, W_hh, gpart);
    k_lstm_pointwise<<<128, 256, 0, stream>>>(gpart, xg + (size_t)t * BAT * G4, cbuf,
                                              hbuf + (size_t)(t + 1) * BAT * HID);
  }

  // Hs rows (t*B+b) start at hbuf + B*H (h_1..h_T).
  if (cvt_big)
    k_gemm_out<__bf16><<<3000, 256, 0, stream>>>(hbuf + (size_t)BAT * HID, Wout_b, b_out, out);
  else
    k_gemm_out<float><<<3000, 256, 0, stream>>>(hbuf + (size_t)BAT * HID, W_out, b_out, out);

  k_argmax<<<1536, 256, 0, stream>>>(out, out + (size_t)SEQ * BAT * VOC);
}
